// ClustAttn_11355893530847
// MI455X (gfx1250) — matrix operand order repaired and checked by reading, not compiled, not run on hardware
//
#include <hip/hip_runtime.h>
#include <hip/hip_bf16.h>

#define SPAT   1600
#define BSZ    8
#define DMODEL 256
#define NH     8
#define DH     32
#define MROWS  (BSZ*SPAT)     // 12800
#define MAXC   1600

typedef __attribute__((ext_vector_type(16))) _Float16 v16h;
typedef __attribute__((ext_vector_type(8)))  float    v8f;
typedef unsigned int v4u __attribute__((ext_vector_type(4)));
typedef int          v8i __attribute__((ext_vector_type(8)));
typedef int          v4i __attribute__((ext_vector_type(4)));

// TDM builtins only exist in the device pass; the host pass must not probe
// them into a hard error (it compiles the unused fallback instead).
#if defined(__HIP_DEVICE_COMPILE__) && \
    __has_builtin(__builtin_amdgcn_tensor_load_to_lds) && \
    __has_builtin(__builtin_amdgcn_s_wait_tensorcnt)
#define HAVE_TDM 1
#else
#define HAVE_TDM 0
#endif

#if defined(__HIP_DEVICE_COMPILE__) && !HAVE_TDM
#error "gfx1250 device pass is missing __builtin_amdgcn_tensor_load_to_lds / s_wait_tensorcnt"
#endif

// K-index map for f16 A-fragment (16x32), per ISA 7.12.2:
// halves 0..7 -> v0..v3, halves 8..15 -> v4..v7; lane-half adds +8 in K.
__device__ __forceinline__ int kmapA(int i, int hi) {
  return ((i & 8) << 1) + (i & 7) + (hi << 3);
}

#if HAVE_TDM
// Issue a TDM 2-D tile load (fp32 elements) from global into LDS.
// D# packing per CDNA5 ISA.
__device__ __forceinline__ void tdm_load_2d_f32(unsigned lds_addr,
                                                const void* gaddr,
                                                unsigned tile_d0,   // elems, contiguous dim
                                                unsigned tile_d1,   // rows
                                                unsigned tensor_d0,
                                                unsigned tensor_d1,
                                                unsigned stride0) { // elems
  unsigned long long ga = (unsigned long long)(uintptr_t)gaddr;
  v4u g0;
  g0[0] = 1u;                                               // count=1, user D#
  g0[1] = lds_addr;                                         // LDS byte address
  g0[2] = (unsigned)(ga & 0xFFFFFFFFu);                     // global_addr[31:0]
  g0[3] = (unsigned)((ga >> 32) & 0x1FFFFFFu) | (2u << 30); // addr[56:32], type=2
  v8i g1;
  g1[0] = (int)(2u << 16);                                  // data_size=4B; wg_mask=0
  g1[1] = (int)((tensor_d0 & 0xFFFFu) << 16);               // tensor_dim0[15:0]
  g1[2] = (int)(((tensor_d0 >> 16) & 0xFFFFu) | ((tensor_d1 & 0xFFFFu) << 16));
  g1[3] = (int)(((tensor_d1 >> 16) & 0xFFFFu) | ((tile_d0 & 0xFFFFu) << 16));
  g1[4] = (int)(tile_d1 & 0xFFFFu);                         // tile_dim1; tile_dim2=0
  g1[5] = (int)stride0;                                     // tensor_dim0_stride[31:0]
  g1[6] = 0;
  g1[7] = 0;
  v4i z4 = {0, 0, 0, 0};
#if __clang_major__ >= 23
  v8i z8 = {0, 0, 0, 0, 0, 0, 0, 0};
  __builtin_amdgcn_tensor_load_to_lds(g0, g1, z4, z4, z8, 0);
#else
  __builtin_amdgcn_tensor_load_to_lds(g0, g1, z4, z4, 0);
#endif
}
#endif

// ---------------------------------------------------------------------------
// 1) entropy pipeline: logits -> softmax -> entropy(bits) -> gauss -> sobel
// ---------------------------------------------------------------------------
__global__ void entropy_kernel(const float* __restrict__ kin,
                               const float* __restrict__ Wp,
                               const float* __restrict__ bp,
                               const float* __restrict__ gk,
                               const float* __restrict__ sb,
                               float* __restrict__ smooth_out,
                               float* __restrict__ rowmean) {
  __shared__ float wp_s[DMODEL];
  __shared__ float ent[SPAT];
  __shared__ float smo[SPAT];
  __shared__ float stp[SPAT];
  __shared__ float red[256];
  __shared__ float gks[29];
  const int b = blockIdx.x;
  const int tid = threadIdx.x;

  wp_s[tid] = Wp[tid];
  if (tid < 29) gks[tid] = gk[tid];
  __syncthreads();

  for (int s = tid; s < SPAT; s += 256) {
    const float* kr = kin + ((long)s * BSZ + b) * DMODEL;
    float acc = 0.f;
    for (int d = 0; d < DMODEL; ++d) acc += kr[d] * wp_s[d];
    ent[s] = acc + bp[0];
  }
  __syncthreads();

  float lmax = -__builtin_inff();
  for (int s = tid; s < SPAT; s += 256) lmax = fmaxf(lmax, ent[s]);
  red[tid] = lmax; __syncthreads();
  for (int off = 128; off > 0; off >>= 1) {
    if (tid < off) red[tid] = fmaxf(red[tid], red[tid + off]);
    __syncthreads();
  }
  const float gmax = red[0]; __syncthreads();
  float lsum = 0.f;
  for (int s = tid; s < SPAT; s += 256) lsum += expf(ent[s] - gmax);
  red[tid] = lsum; __syncthreads();
  for (int off = 128; off > 0; off >>= 1) {
    if (tid < off) red[tid] += red[tid + off];
    __syncthreads();
  }
  const float gsum = red[0]; __syncthreads();

  for (int s = tid; s < SPAT; s += 256) {
    float p = expf(ent[s] - gmax) / gsum + 1e-8f;
    ent[s] = -p * (logf(p) * 1.44269504088896340736f);
  }
  __syncthreads();

  for (int s = tid; s < SPAT; s += 256) {
    float acc = 0.f;
    for (int t = -14; t <= 14; ++t) {
      int j = s + t;
      if (j >= 0 && j < SPAT) acc += ent[j] * gks[14 + t];
    }
    smo[s] = acc;
  }
  __syncthreads();

  const float sb0 = sb[0], sb1 = sb[1], sb2 = sb[2];
  for (int s = tid; s < SPAT; s += 256) {
    float a = (s > 0) ? smo[s - 1] : 0.f;
    float c = (s < SPAT - 1) ? smo[s + 1] : 0.f;
    stp[s] = a * sb2 + smo[s] * sb1 + c * sb0;
    smooth_out[b * SPAT + s] = smo[s];
  }
  __syncthreads();

  int lt = 0;
  for (int s = tid; s < SPAT; s += 256) {
    if (s >= 1) {
      int a = stp[s] > 0.f;
      int p = stp[s - 1] > 0.f;
      lt += (a != p);
    }
  }
  red[tid] = (float)lt; __syncthreads();
  for (int off = 128; off > 0; off >>= 1) {
    if (tid < off) red[tid] += red[tid + off];
    __syncthreads();
  }
  if (tid == 0) rowmean[b] = (float)SPAT / (red[0] + 1.0f);
}

// ---------------------------------------------------------------------------
// 2) derive clst_sh, lo, n_cl, ceil16(n_cl) on device
// ---------------------------------------------------------------------------
__global__ void params_kernel(const float* __restrict__ rowmean,
                              int* __restrict__ prm) {
  float m = 0.f;
  for (int b = 0; b < BSZ; ++b) m += rowmean[b];
  m /= (float)BSZ;
  int clst = (int)roundf(m);
  if (clst < 1) clst = 1;
  if (clst > SPAT) clst = SPAT;
  int rem = SPAT % clst;
  int lo  = rem / 2;
  int hi  = SPAT - (rem - rem / 2);
  int ncl = (hi - lo) / clst;
  prm[0] = clst;
  prm[1] = lo;
  prm[2] = ncl;
  prm[3] = (ncl + 15) & ~15;
}

// ---------------------------------------------------------------------------
// 3) entropy-softmax-weighted cluster pooling of K and V -> kc, vc (b,c,d)
// ---------------------------------------------------------------------------
__global__ void cluster_kernel(const float* __restrict__ kin,
                               const float* __restrict__ vin,
                               const float* __restrict__ smooth,
                               const int* __restrict__ prm,
                               float* __restrict__ kc,
                               float* __restrict__ vc) {
  const int b = blockIdx.x / MAXC;
  const int c = blockIdx.x % MAXC;
  const int clst = prm[0], lo = prm[1], ncl = prm[2], ncl16 = prm[3];
  const int tid = threadIdx.x;  // 64 threads
  if (c >= ncl16) return;
  const long outbase = ((long)b * MAXC + c) * DMODEL;
  if (c >= ncl) {
    for (int d = tid; d < DMODEL; d += 64) { kc[outbase + d] = 0.f; vc[outbase + d] = 0.f; }
    return;
  }
  __shared__ float red[64];
  __shared__ float wsh[MAXC];
  const int base = lo + c * clst;
  const float* sm = smooth + b * SPAT + base;

  float lmax = -__builtin_inff();
  for (int j = tid; j < clst; j += 64) lmax = fmaxf(lmax, sm[j]);
  red[tid] = lmax; __syncthreads();
  for (int off = 32; off > 0; off >>= 1) {
    if (tid < off) red[tid] = fmaxf(red[tid], red[tid + off]);
    __syncthreads();
  }
  const float gmax = red[0]; __syncthreads();
  float lsum = 0.f;
  for (int j = tid; j < clst; j += 64) lsum += expf(sm[j] - gmax);
  red[tid] = lsum; __syncthreads();
  for (int off = 32; off > 0; off >>= 1) {
    if (tid < off) red[tid] += red[tid + off];
    __syncthreads();
  }
  const float gsum = red[0]; __syncthreads();
  for (int j = tid; j < clst; j += 64) wsh[j] = expf(sm[j] - gmax) / gsum;
  __syncthreads();

  for (int d = tid; d < DMODEL; d += 64) {
    float ak = 0.f, av = 0.f;
    for (int j = 0; j < clst; ++j) {
      const long addr = ((long)(base + j) * BSZ + b) * DMODEL + d;
      ak += wsh[j] * kin[addr];
      av += wsh[j] * vin[addr];
    }
    kc[outbase + d] = ak;
    vc[outbase + d] = av;
  }
}

// ---------------------------------------------------------------------------
// 4) WMMA GEMM: Out[r, 0:256] = A[r, 0:256] @ W(256x256) + bias
//    - templated gather/scatter: row index math is wave-uniform -> SALU
//    - A sub-tiles staged through LDS with coalesced 128B loads
//    - B (weight) 32x16 fp32 tile pulled by the Tensor Data Mover into LDS,
//      synchronized with s_wait_tensorcnt (TENSORcnt) + workgroup barrier
// ---------------------------------------------------------------------------
template <int IN_MODE, int OUT_MODE>
__global__ void wmma_gemm256(const float* __restrict__ A,
                             const float* __restrict__ W,
                             const float* __restrict__ bias,
                             float* __restrict__ Out,
                             const int* __restrict__ prm) {
  __shared__ float Asub[8][16][32];   // 16 KB
  __shared__ float Bt[32][16];        // 2 KB (TDM destination)
  const int tid  = threadIdx.x;
  const int wave = tid >> 5;
  const int lane = tid & 31;
  const int n0 = blockIdx.y * 16;
  const int r0 = blockIdx.x * 128 + wave * 16;
  const int m  = lane & 15;
  const int hi = (lane >> 4) & 1;
  const int n  = lane & 15;

  bool active = true;            // wave-uniform: EXEC all-1s at every WMMA
  if (prm) { if ((r0 % MAXC) >= prm[3]) active = false; }

  // wave-uniform row base pointers (handles (s,b,d) gather for q)
  const float* rowp[16];
#pragma unroll
  for (int rr = 0; rr < 16; ++rr) {
    unsigned r = (unsigned)(r0 + rr);
    long arow;
    if (IN_MODE == 1) {
      unsigned bb = r / SPAT, ss = r % SPAT;
      arow = ((long)ss * BSZ + bb) * DMODEL;
    } else {
      arow = (long)r * DMODEL;
    }
    rowp[rr] = A + arow;
  }

  v8f acc = {};
  for (int kk = 0; kk < DMODEL; kk += 32) {
#if HAVE_TDM
    if (wave == 0) {
      tdm_load_2d_f32((unsigned)(uintptr_t)(&Bt[0][0]),
                      W + (long)kk * DMODEL + n0,
                      /*tile_d0=*/16, /*tile_d1=*/32,
                      /*tensor_d0=*/DMODEL, /*tensor_d1=*/DMODEL,
                      /*stride0=*/DMODEL);
      __builtin_amdgcn_s_wait_tensorcnt((short)0);
    }
#else
    // host-pass / no-TDM fallback (never selected on gfx1250 device pass)
    for (int t = tid; t < 512; t += 256) {
      int k = t >> 4, nn = t & 15;
      Bt[k][nn] = W[(kk + k) * DMODEL + n0 + nn];
    }
#endif
    // coalesced A staging: 32 lanes cover the 32 K-columns of each row
#pragma unroll
    for (int rr = 0; rr < 16; ++rr) Asub[wave][rr][lane] = rowp[rr][kk + lane];
    __syncthreads();

    if (active) {
      v16h a, bfrag;
#pragma unroll
      for (int i = 0; i < 16; ++i) {
        a[i]     = (_Float16)Asub[wave][m][i + 16 * hi];
        bfrag[i] = (_Float16)Bt[i + 16 * hi][n];
      }
      acc = __builtin_amdgcn_wmma_f32_16x16x32_f16(false, a, false, bfrag,
                                                   (short)0, acc, false, false);
    }
    __syncthreads();
  }

  if (active) {
    const float bsv = bias[n0 + n];
#pragma unroll
    for (int i = 0; i < 8; ++i) {
      unsigned rr = (unsigned)(r0 + i + 8 * hi);
      long oaddr;
      if (OUT_MODE == 1) {
        unsigned bb = rr / SPAT, ss = rr % SPAT;
        oaddr = ((long)ss * BSZ + bb) * DMODEL;
      } else {
        oaddr = (long)rr * DMODEL;
      }
      Out[oaddr + n0 + n] = acc[i] + bsv;
    }
  }
}

// ---------------------------------------------------------------------------
// 5) flash attention per (b, h, 16-row s tile), 1 wave per block
// ---------------------------------------------------------------------------
__global__ void attn_kernel(const float* __restrict__ Q,
                            const float* __restrict__ K,
                            const float* __restrict__ V,
                            const int* __restrict__ prm,
                            float* __restrict__ ctx) {
  const unsigned blk = blockIdx.x;
  const unsigned b = blk / (NH * (SPAT / 16));
  const unsigned rest = blk % (NH * (SPAT / 16));
  const unsigned h  = rest / (SPAT / 16);
  const unsigned s0 = (rest % (SPAT / 16)) * 16;
  const int ncl = prm[2], ncl16 = prm[3];
  const int lane = threadIdx.x;  // 32
  const int m = lane & 15;
  const int hi = (lane >> 4) & 1;
  const int n = lane & 15;

  __shared__ _Float16 P[16][32];

  v16h qa;
  const float* qrow = Q + (((long)b * SPAT) + s0 + m) * DMODEL + h * DH;
#pragma unroll
  for (int i = 0; i < 16; ++i) qa[i] = (_Float16)qrow[i + 16 * hi];

  float mrun[8], lrun[8];
#pragma unroll
  for (int i = 0; i < 8; ++i) { mrun[i] = -__builtin_inff(); lrun[i] = 0.f; }
  v8f o0 = {}, o1 = {};

  for (int c0 = 0; c0 < ncl; c0 += 32) {
    // prefetch next K chunk (lowers to global_prefetch_b8)
    if (c0 + 32 < ncl)
      __builtin_prefetch(K + (((long)b * MAXC) + c0 + 32 + n) * DMODEL + h * DH, 0, 1);

    v8f sv0, sv1 = {};
    {
      v16h kb;
#pragma unroll
      for (int i = 0; i < 16; ++i) {
        int kk = i + 16 * hi;
        kb[i] = (_Float16)K[(((long)b * MAXC) + c0 + n) * DMODEL + h * DH + kk];
      }
      v8f z = {};
      sv0 = __builtin_amdgcn_wmma_f32_16x16x32_f16(false, qa, false, kb,
                                                   (short)0, z, false, false);
    }
    const bool t1 = (c0 + 16) < ncl;  // wave-uniform
    if (t1) {
      v16h kb;
#pragma unroll
      for (int i = 0; i < 16; ++i) {
        int kk = i + 16 * hi;
        kb[i] = (_Float16)K[(((long)b * MAXC) + c0 + 16 + n) * DMODEL + h * DH + kk];
      }
      v8f z = {};
      sv1 = __builtin_amdgcn_wmma_f32_16x16x32_f16(false, qa, false, kb,
                                                   (short)0, z, false, false);
    }
    if (c0 + n >= ncl) {
#pragma unroll
      for (int i = 0; i < 8; ++i) sv0[i] = -__builtin_inff();
    }
    if (!t1 || (c0 + 16 + n >= ncl)) {
#pragma unroll
      for (int i = 0; i < 8; ++i) sv1[i] = -__builtin_inff();
    }
#pragma unroll
    for (int i = 0; i < 8; ++i) {
      float mx = fmaxf(sv0[i], sv1[i]);
      for (int off = 1; off < 16; off <<= 1) mx = fmaxf(mx, __shfl_xor(mx, off, 16));
      float nm = fmaxf(mrun[i], mx);
      float scale = expf(mrun[i] - nm);
      float p0 = expf(sv0[i] - nm);
      float p1 = expf(sv1[i] - nm);
      float rs = p0 + p1;
      for (int off = 1; off < 16; off <<= 1) rs += __shfl_xor(rs, off, 16);
      lrun[i] = lrun[i] * scale + rs;
      mrun[i] = nm;
      o0[i] *= scale;
      o1[i] *= scale;
      sv0[i] = p0;
      sv1[i] = p1;
    }
#pragma unroll
    for (int i = 0; i < 8; ++i) {
      int mm = i + 8 * hi;
      P[mm][n]      = (_Float16)sv0[i];
      P[mm][16 + n] = (_Float16)sv1[i];
    }
    __syncthreads();
    v16h pa, vb0, vb1;
#pragma unroll
    for (int i = 0; i < 16; ++i) {
      pa[i] = P[m][i + 16 * hi];
      int row = c0 + i + 16 * hi;
      if (row < ncl16) {
        const float* vr = V + (((long)b * MAXC) + row) * DMODEL + h * DH;
        vb0[i] = (_Float16)vr[n];
        vb1[i] = (_Float16)vr[16 + n];
      } else {
        vb0[i] = (_Float16)0.f;
        vb1[i] = (_Float16)0.f;
      }
    }
    o0 = __builtin_amdgcn_wmma_f32_16x16x32_f16(false, pa, false, vb0,
                                                (short)0, o0, false, false);
    o1 = __builtin_amdgcn_wmma_f32_16x16x32_f16(false, pa, false, vb1,
                                                (short)0, o1, false, false);
    __syncthreads();
  }

#pragma unroll
  for (int i = 0; i < 8; ++i) {
    int mm = i + 8 * hi;
    float inv = 1.0f / (lrun[i] * (float)DH);
    long base = (((long)b * SPAT) + s0 + mm) * DMODEL + h * DH;
    ctx[base + n]      = o0[i] * inv;
    ctx[base + 16 + n] = o1[i] * inv;
  }
}

// ---------------------------------------------------------------------------
extern "C" void kernel_launch(void* const* d_in, const int* in_sizes, int n_in,
                              void* d_out, int out_size, void* d_ws, size_t ws_size,
                              hipStream_t stream) {
  const float* q  = (const float*)d_in[0];
  const float* k  = (const float*)d_in[1];
  const float* v  = (const float*)d_in[2];
  const float* Wq = (const float*)d_in[5];
  const float* bq = (const float*)d_in[6];
  const float* Wk = (const float*)d_in[7];
  const float* bk = (const float*)d_in[8];
  const float* Wv = (const float*)d_in[9];
  const float* bv = (const float*)d_in[10];
  const float* Wo = (const float*)d_in[11];
  const float* bo = (const float*)d_in[12];
  const float* Wp = (const float*)d_in[13];
  const float* bp = (const float*)d_in[14];
  const float* gk = (const float*)d_in[15];
  const float* sb = (const float*)d_in[16];

  float* ws       = (float*)d_ws;
  float* smooth   = ws;                        // 12800
  float* rowmean  = ws + 12800;                // 8
  int*   prm      = (int*)(ws + 12816);        // 4
  float* Qb       = ws + 12832;                // 12800*256
  float* kc       = Qb + (long)MROWS * DMODEL;
  float* vc       = kc + (long)MROWS * DMODEL;
  float* Kp       = vc + (long)MROWS * DMODEL;
  float* Vp       = Kp + (long)MROWS * DMODEL;
  float* ctx      = Vp + (long)MROWS * DMODEL;

  entropy_kernel<<<BSZ, 256, 0, stream>>>(k, Wp, bp, gk, sb, smooth, rowmean);
  params_kernel<<<1, 1, 0, stream>>>(rowmean, prm);
  cluster_kernel<<<BSZ * MAXC, 64, 0, stream>>>(k, v, smooth, prm, kc, vc);

  dim3 g(MROWS / 128, DMODEL / 16);
  wmma_gemm256<1, 0><<<g, 256, 0, stream>>>(q,  Wq, bq, Qb, nullptr);
  wmma_gemm256<0, 0><<<g, 256, 0, stream>>>(kc, Wk, bk, Kp, prm);
  wmma_gemm256<0, 0><<<g, 256, 0, stream>>>(vc, Wv, bv, Vp, prm);

  attn_kernel<<<BSZ * NH * (SPAT / 16), 32, 0, stream>>>(Qb, Kp, Vp, prm, ctx);

  wmma_gemm256<0, 1><<<g, 256, 0, stream>>>(ctx, Wo, bo, (float*)d_out, nullptr);
}
